// GAT_44813688766464
// MI455X (gfx1250) — matrix operand order repaired and checked by reading, not compiled, not run on hardware
//
#include <hip/hip_runtime.h>
#include <cstring>
#include <cstdint>

typedef __attribute__((ext_vector_type(16))) __bf16 v16bf;
typedef __attribute__((ext_vector_type(8)))  float  v8f;

#define NEG_SLOPE 0.2f
#define BN_EPS    1e-5f
#define MT        4      // M sub-tiles per wave (64 rows)

// ---------------- helpers ----------------
__device__ __forceinline__ unsigned d_encf(float f) {
  unsigned u = __float_as_uint(f);
  return (u & 0x80000000u) ? ~u : (u | 0x80000000u);
}
__device__ __forceinline__ float d_decf(unsigned u) {
  u = (u & 0x80000000u) ? (u & 0x7FFFFFFFu) : ~u;
  return __uint_as_float(u);
}
static unsigned h_encf(float f) {
  unsigned u; memcpy(&u, &f, 4);
  return (u & 0x80000000u) ? ~u : (u | 0x80000000u);
}
static inline int ceil_div(int a, int b) { return (a + b - 1) / b; }

// ---------------- WMMA GEMM: C = act(A[M,K] @ B[K,N] + bias) ----------------
// bf16 inputs (converted in-register from fp32), fp32 accumulation.
// One wave -> 64x16 output (4 stacked 16x16 WMMA tiles). The 32x64 fp32 B panel
// for the block is staged cooperatively into LDS with CDNA5 async global->LDS
// DMA (GLOBAL_LOAD_ASYNC_TO_LDS_B128, ASYNCcnt-tracked), then consumed via LDS
// reads; A is loaded directly as b128 and fed to v_wmma_f32_16x16x32_bf16.
// Requires N%64==0, K%32==0, M%16==0 (tail M sub-tiles skipped wave-uniformly).
__global__ __launch_bounds__(128)
void k_wmma_gemm(const float* __restrict__ A, const float* __restrict__ B,
                 const float* __restrict__ bias, float* __restrict__ C,
                 int M, int Nn, int Kk, int act)
{
  __shared__ float Bs[32 * 64];            // one K-chunk of the block's B panel (8 KB)

  const int wave = threadIdx.x >> 5;
  const int lane = threadIdx.x & 31;
  const int half = lane >> 4;              // which 16-lane half of the wave
  const int l15  = lane & 15;
  const int nbase = blockIdx.y * 64;       // block's N window (4 waves x 16)
  const int tileN = nbase + wave * 16;
  const int baseM = blockIdx.x * (16 * MT);
  if (baseM >= M) return;                  // block-uniform: no barrier divergence

  // async-copy coordinates: one b128 op moves 2 K-rows (32 lanes * 16B = 512B)
  const int colq = l15 * 4;                                 // 4-float column quarter
  const unsigned lds_base = (unsigned)(uintptr_t)Bs;        // LDS offset (low 32 bits)

  v8f acc[MT];
#pragma unroll
  for (int mi = 0; mi < MT; ++mi) acc[mi] = (v8f){};

  for (int k0 = 0; k0 < Kk; k0 += 32) {
    // ---- stage B[k0..k0+32) x [nbase..nbase+64) into LDS (wave w: K-rows w*8..w*8+8) ----
#pragma unroll
    for (int i = 0; i < 4; ++i) {
      const int kr = wave * 8 + 2 * i + half;               // chunk-local K row
      const float* g = B + (size_t)(k0 + kr) * Nn + nbase + colq;
      const unsigned loff = lds_base + (unsigned)((kr * 64 + colq) * 4);
      asm volatile("global_load_async_to_lds_b128 %0, %1, off"
                   :: "v"(loff), "v"(g) : "memory");
    }
    asm volatile("s_wait_asynccnt 0" ::: "memory");
    __syncthreads();                                        // panel visible to all waves

    // ---- B fragment from LDS: lane col = lane&15, K = half*8 + e (+8 for e>=8), matching A ----
    v16bf bv;
#pragma unroll
    for (int e = 0; e < 16; ++e)
      bv[e] = (__bf16)Bs[(half * 8 + e + (e >= 8 ? 8 : 0)) * 64 + wave * 16 + l15];

#pragma unroll
    for (int mi = 0; mi < MT; ++mi) {
      if (baseM + mi * 16 >= M) break;     // wave-uniform guard: EXEC stays all-1s
      // --- A tile 16x32 bf16, ISA layout: lanes<16 hold K=[0..8)&[16..24), lanes>=16 +8 ---
      const float* ap = A + (size_t)(baseM + mi * 16 + l15) * Kk + k0 + half * 8;
      __builtin_prefetch(ap + 32, 0, 3);
      float af[16];
      *(float4*)(af + 0)  = *(const float4*)(ap + 0);
      *(float4*)(af + 4)  = *(const float4*)(ap + 4);
      *(float4*)(af + 8)  = *(const float4*)(ap + 16);
      *(float4*)(af + 12) = *(const float4*)(ap + 20);
      v16bf av;
#pragma unroll
      for (int e = 0; e < 16; ++e) av[e] = (__bf16)af[e];

      acc[mi] = __builtin_amdgcn_wmma_f32_16x16x32_bf16(
          /*neg_a=*/false, av, /*neg_b=*/false, bv,
          /*c_mod=*/(short)0, acc[mi], /*reuse_a=*/false, /*reuse_b=*/false);
    }
    __syncthreads();                       // all reads done before next panel overwrite
  }

  const float bb = bias ? bias[tileN + l15] : 0.f;
#pragma unroll
  for (int mi = 0; mi < MT; ++mi) {
    if (baseM + mi * 16 >= M) break;
#pragma unroll
    for (int r = 0; r < 8; ++r) {
      const int row = baseM + mi * 16 + half * 8 + r;  // C/D layout: lanes>=16 hold M=8..15
      float v = acc[mi][r] + bb;
      if (act) v = tanhf(v);
      C[(size_t)row * Nn + tileN + l15] = v;
    }
  }
}

// ---------------- per-(node,head) attention dot: out[i,h] = sum_c xs[i,h,c]*att[h,c] ----------------
__global__ __launch_bounds__(128)
void k_rowdot_att(const float* __restrict__ xs, const float* __restrict__ att,
                  float* __restrict__ out, int rows, int H, int C)
{
  const int wid  = blockIdx.x * 4 + (threadIdx.x >> 5);
  if (wid >= rows * H) return;
  const int lane = threadIdx.x & 31;
  const int node = wid / H, h = wid % H;
  const float* xr = xs + (size_t)node * H * C + (size_t)h * C;
  const float* ar = att + (size_t)h * C;
  float p = 0.f;
  for (int c = lane; c < C; c += 32) p += xr[c] * ar[c];
#pragma unroll
  for (int off = 16; off; off >>= 1) p += __shfl_xor(p, off, 32);
  if (lane == 0) out[wid] = p;
}

// ---------------- fills ----------------
__global__ void k_fill_u32(unsigned* p, unsigned v, int n) {
  int i = blockIdx.x * blockDim.x + threadIdx.x;
  if (i < n) p[i] = v;
}

// ---------------- edge list builders ----------------
__global__ void k_build_edges1(const int* __restrict__ eidx, int E, int n,
                               int* __restrict__ es, int* __restrict__ ed)
{
  int e = blockIdx.x * blockDim.x + threadIdx.x;
  if (e >= E + n) return;
  if (e < E) { es[e] = eidx[e]; ed[e] = eidx[E + e]; }
  else       { es[e] = ed[e] = e - E; }
}
__global__ void k_build_edges2(const int* __restrict__ eidx, int E, int k,
                               const int* __restrict__ new_idx,
                               int* __restrict__ es, int* __restrict__ ed)
{
  int e = blockIdx.x * blockDim.x + threadIdx.x;
  if (e >= E + k) return;
  if (e < E) {
    int s = new_idx[eidx[e]], d = new_idx[eidx[E + e]];
    if (s < 0 || d < 0) { es[e] = -1; ed[e] = 0; }
    else                { es[e] = s;  ed[e] = d; }
  } else { es[e] = ed[e] = e - E; }
}

// ---------------- segment softmax passes ----------------
__global__ void k_edge_max(const int* __restrict__ es, const int* __restrict__ ed, int nE,
                           const float* __restrict__ as_, const float* __restrict__ ad_,
                           unsigned* __restrict__ mx, int H)
{
  int e = blockIdx.x * blockDim.x + threadIdx.x;
  if (e >= nE) return;
  int s = es[e]; if (s < 0) return;
  int d = ed[e];
  for (int h = 0; h < H; ++h) {
    float l = as_[s * H + h] + ad_[d * H + h];
    l = l > 0.f ? l : NEG_SLOPE * l;
    atomicMax(&mx[d * H + h], d_encf(l));
  }
}
__global__ void k_edge_sum(const int* __restrict__ es, const int* __restrict__ ed, int nE,
                           const float* __restrict__ as_, const float* __restrict__ ad_,
                           const unsigned* __restrict__ mx, float* __restrict__ sm, int H)
{
  int e = blockIdx.x * blockDim.x + threadIdx.x;
  if (e >= nE) return;
  int s = es[e]; if (s < 0) return;
  int d = ed[e];
  for (int h = 0; h < H; ++h) {
    float l = as_[s * H + h] + ad_[d * H + h];
    l = l > 0.f ? l : NEG_SLOPE * l;
    atomicAdd(&sm[d * H + h], __expf(l - d_decf(mx[d * H + h])));
  }
}
// wave-per-edge message scatter: acc[d, h, c] += xs[s, h, c] * alpha[e, h]
__global__ __launch_bounds__(128)
void k_edge_msg(const int* __restrict__ es, const int* __restrict__ ed, int nE,
                const float* __restrict__ as_, const float* __restrict__ ad_,
                const unsigned* __restrict__ mx, const float* __restrict__ sm,
                const float* __restrict__ xs, float* __restrict__ acc, int C /*per head; H==2*/)
{
  const int wid = blockIdx.x * 4 + (threadIdx.x >> 5);
  if (wid >= nE) return;
  const int s = es[wid]; if (s < 0) return;
  const int d = ed[wid];
  const int lane = threadIdx.x & 31;

  float alpha0, alpha1;
  {
    float l0 = as_[s * 2 + 0] + ad_[d * 2 + 0]; l0 = l0 > 0.f ? l0 : NEG_SLOPE * l0;
    float l1 = as_[s * 2 + 1] + ad_[d * 2 + 1]; l1 = l1 > 0.f ? l1 : NEG_SLOPE * l1;
    alpha0 = __expf(l0 - d_decf(mx[d * 2 + 0])) / sm[d * 2 + 0];
    alpha1 = __expf(l1 - d_decf(mx[d * 2 + 1])) / sm[d * 2 + 1];
  }
  const int HC = 2 * C;
  const float* xr = xs  + (size_t)s * HC;
  float*       orw = acc + (size_t)d * HC;
  for (int c = lane; c < HC; c += 32)
    atomicAdd(&orw[c], xr[c] * (c < C ? alpha0 : alpha1));
}

// ---------------- bias + tanh ----------------
__global__ void k_bias_tanh(float* __restrict__ acc, const float* __restrict__ bias,
                            int rows, int C)
{
  long long i = (long long)blockIdx.x * blockDim.x + threadIdx.x;
  if (i >= (long long)rows * C) return;
  int c = (int)(i % C);
  acc[i] = tanhf(acc[i] + bias[c]);
}

// ---------------- SAGPool: add-aggregation, score, rank-topk, gather ----------------
__global__ __launch_bounds__(128)
void k_edge_agg(const int* __restrict__ es, const int* __restrict__ ed, int nE,
                const float* __restrict__ x1, float* __restrict__ agg, int C)
{
  const int wid = blockIdx.x * 4 + (threadIdx.x >> 5);
  if (wid >= nE) return;
  const int s = es[wid], d = ed[wid];
  const int lane = threadIdx.x & 31;
  const float* xr = x1 + (size_t)s * C;
  float* ar = agg + (size_t)d * C;
  for (int c = lane; c < C; c += 32) atomicAdd(&ar[c], xr[c]);
}
__global__ __launch_bounds__(128)
void k_score(const float* __restrict__ agg, const float* __restrict__ x1,
             const float* __restrict__ w_rel, const float* __restrict__ w_root,
             const float* __restrict__ b_sag, float* __restrict__ score, int n, int C)
{
  const int wid = blockIdx.x * 4 + (threadIdx.x >> 5);
  if (wid >= n) return;
  const int lane = threadIdx.x & 31;
  const float* ar = agg + (size_t)wid * C;
  const float* xr = x1 + (size_t)wid * C;
  float p = 0.f;
  for (int c = lane; c < C; c += 32) p += ar[c] * w_rel[c] + xr[c] * w_root[c];
#pragma unroll
  for (int off = 16; off; off >>= 1) p += __shfl_xor(p, off, 32);
  if (lane == 0) score[wid] = tanhf(p + b_sag[0]);
}
__global__ __launch_bounds__(256)
void k_rank(const float* __restrict__ score, int n, int k,
            int* __restrict__ new_idx, int* __restrict__ perm, float* __restrict__ vals)
{
  __shared__ int sh[256];
  const int i = blockIdx.x;
  const float si = score[i];
  int cnt = 0;
  for (int j = threadIdx.x; j < n; j += 256) {
    float sj = score[j];
    cnt += (sj > si) || (sj == si && j < i);
  }
  sh[threadIdx.x] = cnt;
  __syncthreads();
  for (int s = 128; s; s >>= 1) {
    if (threadIdx.x < s) sh[threadIdx.x] += sh[threadIdx.x + s];
    __syncthreads();
  }
  if (threadIdx.x == 0) {
    int r = sh[0];
    new_idx[i] = (r < k) ? r : -1;
    if (r < k) { perm[r] = i; vals[r] = si; }
  }
}
__global__ void k_gather_scale(const float* __restrict__ x1, const int* __restrict__ perm,
                               const float* __restrict__ vals, float* __restrict__ xp,
                               int k, int C)
{
  long long i = (long long)blockIdx.x * blockDim.x + threadIdx.x;
  if (i >= (long long)k * C) return;
  int r = (int)(i / C), c = (int)(i % C);
  xp[i] = x1[(size_t)perm[r] * C + c] * vals[r];
}

// ---------------- batchnorm (over rows) ----------------
__global__ __launch_bounds__(256)
void k_colstats(const float* __restrict__ x, int rows, int cols,
                float* __restrict__ mean, float* __restrict__ var)
{
  __shared__ float s1[256], s2[256];
  const int c = blockIdx.x;
  float a = 0.f, b = 0.f;
  for (int r = threadIdx.x; r < rows; r += 256) {
    float v = x[(size_t)r * cols + c];
    a += v; b += v * v;
  }
  s1[threadIdx.x] = a; s2[threadIdx.x] = b;
  __syncthreads();
  for (int s = 128; s; s >>= 1) {
    if (threadIdx.x < s) { s1[threadIdx.x] += s1[threadIdx.x + s]; s2[threadIdx.x] += s2[threadIdx.x + s]; }
    __syncthreads();
  }
  if (threadIdx.x == 0) {
    float m = s1[0] / rows;
    mean[c] = m;
    var[c] = s2[0] / rows - m * m;
  }
}
__global__ void k_bn_apply(float* __restrict__ x, const float* __restrict__ mean,
                           const float* __restrict__ var, const float* __restrict__ g,
                           const float* __restrict__ b, int rows, int cols, int act)
{
  long long i = (long long)blockIdx.x * blockDim.x + threadIdx.x;
  if (i >= (long long)rows * cols) return;
  int c = (int)(i % cols);
  float v = (x[i] - mean[c]) * rsqrtf(var[c] + BN_EPS) * g[c] + b[c];
  if (act) v = tanhf(v);
  x[i] = v;
}

// ---------------- fc2 (N=3) + log_softmax ----------------
__global__ __launch_bounds__(128)
void k_fc2_lsm(const float* __restrict__ h, const float* __restrict__ w,
               const float* __restrict__ b, float* __restrict__ out, int rows, int K)
{
  const int wid = blockIdx.x * 4 + (threadIdx.x >> 5);
  if (wid >= rows) return;
  const int lane = threadIdx.x & 31;
  const float* hr = h + (size_t)wid * K;
  float d0 = 0.f, d1 = 0.f, d2 = 0.f;
  for (int c = lane; c < K; c += 32) {
    float v = hr[c];
    d0 += v * w[c * 3 + 0];
    d1 += v * w[c * 3 + 1];
    d2 += v * w[c * 3 + 2];
  }
#pragma unroll
  for (int off = 16; off; off >>= 1) {
    d0 += __shfl_xor(d0, off, 32);
    d1 += __shfl_xor(d1, off, 32);
    d2 += __shfl_xor(d2, off, 32);
  }
  if (lane == 0) {
    d0 += b[0]; d1 += b[1]; d2 += b[2];
    float m = fmaxf(d0, fmaxf(d1, d2));
    float lse = m + __logf(__expf(d0 - m) + __expf(d1 - m) + __expf(d2 - m));
    out[wid * 3 + 0] = d0 - lse;
    out[wid * 3 + 1] = d1 - lse;
    out[wid * 3 + 2] = d2 - lse;
  }
}

// ==================== host orchestration ====================
extern "C" void kernel_launch(void* const* d_in, const int* in_sizes, int n_in,
                              void* d_out, int out_size, void* d_ws, size_t ws_size,
                              hipStream_t stream)
{
  const float* x     = (const float*)d_in[0];
  const float* W1s   = (const float*)d_in[1];
  const float* W1d   = (const float*)d_in[2];
  const float* att1s = (const float*)d_in[3];
  const float* att1d = (const float*)d_in[4];
  const float* b1    = (const float*)d_in[5];
  const float* w_rel = (const float*)d_in[6];
  const float* w_root= (const float*)d_in[7];
  const float* b_sag = (const float*)d_in[8];
  const float* W2s   = (const float*)d_in[9];
  const float* W2d   = (const float*)d_in[10];
  const float* att2s = (const float*)d_in[11];
  const float* att2d = (const float*)d_in[12];
  const float* b2    = (const float*)d_in[13];
  const float* g2    = (const float*)d_in[14];
  const float* bt2   = (const float*)d_in[15];
  const float* fc1w  = (const float*)d_in[16];
  const float* fc1b  = (const float*)d_in[17];
  const float* bn1g  = (const float*)d_in[18];
  const float* bn1b  = (const float*)d_in[19];
  const float* fc2w  = (const float*)d_in[20];
  const float* fc2b  = (const float*)d_in[21];
  const int*   eidx  = (const int*)d_in[22];

  const int DIN  = 256;
  const int N    = in_sizes[0] / DIN;    // 20000
  const int C1   = in_sizes[5];          // 512  (H*HID)
  const int H    = 2;
  const int HID  = C1 / H;               // 256
  const int C2   = in_sizes[13];         // 1024 (H*512)
  const int HID2 = C2 / H;               // 512
  const int F1   = in_sizes[17];         // 128
  const int E    = in_sizes[22] / 2;     // 320000
  const int Kp   = (4 * N + 4) / 5;      // ceil(0.8*N) = 16000
  (void)n_in; (void)out_size; (void)ws_size;

  // ---- bump allocator in d_ws ----
  char* wsb = (char*)d_ws;
  size_t off = 0;
  auto alloc = [&](size_t bytes) -> void* {
    void* p = wsb + off;
    off += (bytes + 255) & ~(size_t)255;
    return p;
  };
  float*    R1    = (float*)alloc((size_t)N * C1 * 4);   // xs1, later agg
  float*    R2    = (float*)alloc((size_t)N * C1 * 4);   // xd1, later xp
  float*    R3    = (float*)alloc((size_t)N * C1 * 4);   // acc1 -> x1, later hpre
  float*    as1   = (float*)alloc((size_t)N * H * 4);
  float*    ad1   = (float*)alloc((size_t)N * H * 4);
  unsigned* mx    = (unsigned*)alloc((size_t)N * H * 4); // reused for conv2 (Kp<=N)
  float*    sm    = (float*)alloc((size_t)N * H * 4);    // reused for conv2
  int*      es1   = (int*)alloc((size_t)(E + N) * 4);
  int*      ed1   = (int*)alloc((size_t)(E + N) * 4);
  float*    score = (float*)alloc((size_t)N * 4);
  int*      nidx  = (int*)alloc((size_t)N * 4);
  int*      perm  = (int*)alloc((size_t)Kp * 4);
  float*    vals  = (float*)alloc((size_t)Kp * 4);
  int*      es2   = (int*)alloc((size_t)(E + Kp) * 4);
  int*      ed2   = (int*)alloc((size_t)(E + Kp) * 4);
  float*    as2   = (float*)alloc((size_t)Kp * H * 4);
  float*    ad2   = (float*)alloc((size_t)Kp * H * 4);
  float*    xs2   = (float*)alloc((size_t)Kp * C2 * 4);
  float*    xd2   = (float*)alloc((size_t)Kp * C2 * 4);
  float*    acc2  = (float*)alloc((size_t)Kp * C2 * 4);  // -> x2
  float*    mu2   = (float*)alloc((size_t)C2 * 4);
  float*    var2  = (float*)alloc((size_t)C2 * 4);
  float*    mu1   = (float*)alloc((size_t)F1 * 4);
  float*    var1  = (float*)alloc((size_t)F1 * 4);

  float* xs1  = R1;
  float* xd1  = R2;
  float* acc1 = R3;          // becomes x1 after bias+tanh
  float* agg  = R1;          // reuse after conv1 message pass
  float* xp   = R2;          // reuse
  float* hpre = R3;          // reuse after xp gather

  const unsigned NEG_INF_ENC = h_encf(-1e30f);
  const int TB = 256;
  const int MROWS = 16 * MT; // 64 rows per block in M

  // ===== conv1: linear maps =====
  k_wmma_gemm<<<dim3(ceil_div(N, MROWS), C1 / 64), 128, 0, stream>>>(x, W1s, nullptr, xs1, N, C1, DIN, 0);
  k_wmma_gemm<<<dim3(ceil_div(N, MROWS), C1 / 64), 128, 0, stream>>>(x, W1d, nullptr, xd1, N, C1, DIN, 0);
  k_rowdot_att<<<ceil_div(N * H, 4), 128, 0, stream>>>(xs1, att1s, as1, N, H, HID);
  k_rowdot_att<<<ceil_div(N * H, 4), 128, 0, stream>>>(xd1, att1d, ad1, N, H, HID);

  // ===== conv1: edges (with self loops), segment softmax, scatter =====
  const int nE1 = E + N;
  k_build_edges1<<<ceil_div(nE1, TB), TB, 0, stream>>>(eidx, E, N, es1, ed1);
  k_fill_u32<<<ceil_div(N * H, TB), TB, 0, stream>>>(mx, NEG_INF_ENC, N * H);
  k_fill_u32<<<ceil_div(N * H, TB), TB, 0, stream>>>((unsigned*)sm, 0u, N * H);
  k_edge_max<<<ceil_div(nE1, TB), TB, 0, stream>>>(es1, ed1, nE1, as1, ad1, mx, H);
  k_edge_sum<<<ceil_div(nE1, TB), TB, 0, stream>>>(es1, ed1, nE1, as1, ad1, mx, sm, H);
  k_fill_u32<<<ceil_div(N * C1, TB), TB, 0, stream>>>((unsigned*)acc1, 0u, N * C1);
  k_edge_msg<<<ceil_div(nE1, 4), 128, 0, stream>>>(es1, ed1, nE1, as1, ad1, mx, sm, xs1, acc1, HID);
  k_bias_tanh<<<ceil_div(N * C1, TB), TB, 0, stream>>>(acc1, b1, N, C1);   // acc1 = x1

  // ===== SAGPool =====
  k_fill_u32<<<ceil_div(N * C1, TB), TB, 0, stream>>>((unsigned*)agg, 0u, N * C1);
  k_edge_agg<<<ceil_div(E, 4), 128, 0, stream>>>(es1, ed1, E, acc1, agg, C1);  // no self loops
  k_score<<<ceil_div(N, 4), 128, 0, stream>>>(agg, acc1, w_rel, w_root, b_sag, score, N, C1);
  k_rank<<<N, 256, 0, stream>>>(score, N, Kp, nidx, perm, vals);
  k_gather_scale<<<ceil_div(Kp * C1, TB), TB, 0, stream>>>(acc1, perm, vals, xp, Kp, C1);

  // ===== conv2 =====
  const int nE2 = E + Kp;
  k_build_edges2<<<ceil_div(nE2, TB), TB, 0, stream>>>(eidx, E, Kp, nidx, es2, ed2);
  k_wmma_gemm<<<dim3(ceil_div(Kp, MROWS), C2 / 64), 128, 0, stream>>>(xp, W2s, nullptr, xs2, Kp, C2, C1, 0);
  k_wmma_gemm<<<dim3(ceil_div(Kp, MROWS), C2 / 64), 128, 0, stream>>>(xp, W2d, nullptr, xd2, Kp, C2, C1, 0);
  k_rowdot_att<<<ceil_div(Kp * H, 4), 128, 0, stream>>>(xs2, att2s, as2, Kp, H, HID2);
  k_rowdot_att<<<ceil_div(Kp * H, 4), 128, 0, stream>>>(xd2, att2d, ad2, Kp, H, HID2);
  k_fill_u32<<<ceil_div(Kp * H, TB), TB, 0, stream>>>(mx, NEG_INF_ENC, Kp * H);
  k_fill_u32<<<ceil_div(Kp * H, TB), TB, 0, stream>>>((unsigned*)sm, 0u, Kp * H);
  k_edge_max<<<ceil_div(nE2, TB), TB, 0, stream>>>(es2, ed2, nE2, as2, ad2, mx, H);
  k_edge_sum<<<ceil_div(nE2, TB), TB, 0, stream>>>(es2, ed2, nE2, as2, ad2, mx, sm, H);
  k_fill_u32<<<ceil_div(Kp * C2, TB), TB, 0, stream>>>((unsigned*)acc2, 0u, Kp * C2);
  k_edge_msg<<<ceil_div(nE2, 4), 128, 0, stream>>>(es2, ed2, nE2, as2, ad2, mx, sm, xs2, acc2, HID2);
  k_bias_tanh<<<ceil_div(Kp * C2, TB), TB, 0, stream>>>(acc2, b2, Kp, C2);  // acc2 = x2

  // batchnorm on x2
  k_colstats<<<C2, 256, 0, stream>>>(acc2, Kp, C2, mu2, var2);
  k_bn_apply<<<ceil_div(Kp * C2, TB), TB, 0, stream>>>(acc2, mu2, var2, g2, bt2, Kp, C2, 0);

  // ===== MLP head =====
  k_wmma_gemm<<<dim3(ceil_div(Kp, MROWS), F1 / 64), 128, 0, stream>>>(acc2, fc1w, fc1b, hpre, Kp, F1, C2, 0);
  k_colstats<<<F1, 256, 0, stream>>>(hpre, Kp, F1, mu1, var1);
  k_bn_apply<<<ceil_div(Kp * F1, TB), TB, 0, stream>>>(hpre, mu1, var1, bn1g, bn1b, Kp, F1, 1);
  k_fc2_lsm<<<ceil_div(Kp, 4), 128, 0, stream>>>(hpre, fc2w, fc2b, (float*)d_out, Kp, F1);
}
